// QueryTower_21801253995157
// MI455X (gfx1250) — compile-verified, matrix operand order checked by reading, not run on hardware
//
#include <hip/hip_runtime.h>
#include <hip/hip_bf16.h>

// ---------------------------------------------------------------------------
// QueryTower: 2-layer bidirectional GRU, bf16 WMMA implementation for gfx1250
// B=128, T=64, EMB=200 (padded 224), H=256, 3H=768
// ---------------------------------------------------------------------------

#define Bn   128
#define Tn   64
#define EMB  200
#define KP0  224          // EMB padded to multiple of 32
#define HID  256
#define N3H  768
#define K1   512
#define VOCAB 100000

typedef __attribute__((ext_vector_type(16))) __bf16 v16bf;
typedef __attribute__((ext_vector_type(8)))  __bf16 v8bf;
typedef __attribute__((ext_vector_type(8)))  float  v8f;
typedef __attribute__((ext_vector_type(4)))  int    v4i;

// ---- gfx1250 async global->LDS staging (GLOBAL_LOAD_ASYNC_TO_LDS_B128) -----
#if defined(__has_builtin)
#if __has_builtin(__builtin_amdgcn_global_load_async_to_lds_b128)
#define HAVE_ASYNC_LDS 1
#endif
#endif
#ifndef HAVE_ASYNC_LDS
#define HAVE_ASYNC_LDS 0
#endif

#if HAVE_ASYNC_LDS
typedef __attribute__((address_space(1))) v4i* gas_v4i_ptr;
typedef __attribute__((address_space(3))) v4i* las_v4i_ptr;

__device__ __forceinline__ void async_copy16(const __bf16* g, __bf16* l) {
  __builtin_amdgcn_global_load_async_to_lds_b128(
      (gas_v4i_ptr)(g), (las_v4i_ptr)(l), 0, 0);
}
__device__ __forceinline__ void async_wait0() {
#if __has_builtin(__builtin_amdgcn_s_wait_asynccnt)
  __builtin_amdgcn_s_wait_asynccnt(0);
#else
  asm volatile("s_wait_asynccnt 0x0" ::: "memory");
#endif
}
#endif

// Load a 16-element bf16 WMMA fragment for one lane.
// Layout (A 16x32 bf16, and symmetric for B as [N,K] row-major):
//   lane row = (lane&15); per-lane K chunks at kb and kb+16, kb = ktile*32 + (lane>>4)*8
__device__ __forceinline__ v16bf load_frag(const __bf16* base, int ld, int row, int kb) {
  const __bf16* p = base + (size_t)row * ld + kb;
  v8bf lo = *(const v8bf*)p;
  v8bf hi = *(const v8bf*)(p + 16);
  return __builtin_shufflevector(lo, hi, 0,1,2,3,4,5,6,7,8,9,10,11,12,13,14,15);
}

__device__ __forceinline__ v8f wmma_bf16(v16bf a, v16bf b, v8f c) {
  return __builtin_amdgcn_wmma_f32_16x16x32_bf16(false, a, false, b, (short)0, c, false, false);
}

// ---------------------------------------------------------------------------
// Embedding gather + OOV mask, zero-pad K to 224, emit bf16.  grid = T*B rows.
// ---------------------------------------------------------------------------
__global__ __launch_bounds__(KP0) void embed_pad_kernel(
    const int* __restrict__ idx, const float* __restrict__ emb,
    const float* __restrict__ oov, __bf16* __restrict__ x) {
  const int r = blockIdx.x;            // r = t*B + b
  const int t = r >> 7, b = r & 127;
  const int wi = idx[b * Tn + t];
  const bool isoov = (wi >= VOCAB);
  const float* src = isoov ? oov : (emb + (size_t)wi * EMB);
  const int k = threadIdx.x;
  float v = (k < EMB) ? src[k] : 0.0f;
  x[(size_t)r * KP0 + k] = (__bf16)v;
}

// ---------------------------------------------------------------------------
// f32 [N,K] -> bf16 [N,KP] (zero K-padding)
// ---------------------------------------------------------------------------
__global__ __launch_bounds__(256) void cvt_pad_bf16_kernel(
    const float* __restrict__ src, __bf16* __restrict__ dst,
    int N, int K, int KP) {
  size_t i = (size_t)blockIdx.x * 256 + threadIdx.x;
  if (i >= (size_t)N * KP) return;
  int r = (int)(i / KP), c = (int)(i % KP);
  dst[i] = (__bf16)((c < K) ? src[(size_t)r * K + c] : 0.0f);
}

// ---------------------------------------------------------------------------
// C[M,N] = A[M,K](bf16) * W[N,K](bf16)^T + bias[N]   (f32 out)
// block = 256 threads (8 waves); tile 128x128; K step 32.
// Tile staging uses GLOBAL_LOAD_ASYNC_TO_LDS_B128 when available.
// B-fragments batch-loaded so ds_loads pipeline before the WMMA clause.
// ---------------------------------------------------------------------------
__global__ __launch_bounds__(256) void gemm_bf16_nt_kernel(
    const __bf16* __restrict__ A, const __bf16* __restrict__ Bw,
    const float* __restrict__ bias, float* __restrict__ C,
    int M, int N, int K) {
  __shared__ __bf16 As[128 * 32];
  __shared__ __bf16 Bs[128 * 32];
  const int tid  = threadIdx.x;
  const int wave = tid >> 5, lane = tid & 31;
  const int mBase = blockIdx.y * 128, nBase = blockIdx.x * 128;
  const int lrow = lane & 15;
  const int kb   = (lane >> 4) << 3;      // 0 or 8

  v8f acc[8];
  #pragma unroll
  for (int nt = 0; nt < 8; ++nt)
    #pragma unroll
    for (int e = 0; e < 8; ++e) acc[nt][e] = 0.0f;

  for (int k0 = 0; k0 < K; k0 += 32) {
    __syncthreads();                      // prior-iter fragment reads complete
    #pragma unroll
    for (int h2 = 0; h2 < 2; ++h2) {
      int ch = tid + h2 * 256;            // 512 16B-chunks per tile
      int r  = ch >> 2, c8 = (ch & 3) * 8;
      const __bf16* ga = A  + (size_t)(mBase + r) * K + k0 + c8;
      const __bf16* gb = Bw + (size_t)(nBase + r) * K + k0 + c8;
#if HAVE_ASYNC_LDS
      async_copy16(ga, As + r * 32 + c8);
      async_copy16(gb, Bs + r * 32 + c8);
#else
      *(v8bf*)(As + r * 32 + c8) = *(const v8bf*)ga;
      *(v8bf*)(Bs + r * 32 + c8) = *(const v8bf*)gb;
#endif
      if (k0 + 32 < K) {                  // gfx1250 global_prefetch_b8 hint
        __builtin_prefetch(ga + 32, 0, 1);
        __builtin_prefetch(gb + 32, 0, 1);
      }
    }
#if HAVE_ASYNC_LDS
    async_wait0();                        // ASYNCcnt==0 before barrier
#endif
    __syncthreads();

    v16bf af = load_frag(As, 32, wave * 16 + lrow, kb);
    v16bf bf[8];
    #pragma unroll
    for (int nt = 0; nt < 8; ++nt)        // batch loads -> pipelined ds_loads
      bf[nt] = load_frag(Bs, 32, nt * 16 + lrow, kb);
    #pragma unroll
    for (int nt = 0; nt < 8; ++nt)        // dense WMMA clause
      acc[nt] = wmma_bf16(af, bf[nt], acc[nt]);
  }

  #pragma unroll
  for (int nt = 0; nt < 8; ++nt) {
    int col = nBase + nt * 16 + lrow;
    float bv = bias[col];
    #pragma unroll
    for (int e = 0; e < 8; ++e) {
      int row = mBase + wave * 16 + e + ((lane >> 4) << 3);
      C[(size_t)row * N + col] = acc[nt][e] + bv;
    }
  }
}

// ---------------------------------------------------------------------------
// GRU scan: one block per direction (grid=2), 512 threads = 16 waves.
// Wave w owns hidden-column tile c0 = w*16 -> computes its r/z/n gate tiles.
// h ping-pongs in LDS (bf16), Whh streamed bf16 from L2 per step.
// write_all=1: emit h_t each step into out[T*B,512] at cols dir*256 (layer 0).
// write_all=0: emit only the final state (layer 1).
// ---------------------------------------------------------------------------
__global__ __launch_bounds__(512) void gru_scan_kernel(
    const float* __restrict__ xgF, const float* __restrict__ xgB,
    const __bf16* __restrict__ WhhF, const __bf16* __restrict__ WhhB,
    const float* __restrict__ bhhF, const float* __restrict__ bhhB,
    __bf16* __restrict__ out, int write_all) {
  __shared__ __bf16 h[2][Bn][HID];       // 2 * 128 * 256 * 2B = 128 KB

  const int dir = blockIdx.x;
  const float*  xg  = dir ? xgB  : xgF;
  const __bf16* Whh = dir ? WhhB : WhhF;
  const float*  bhh = dir ? bhhB : bhhF;

  const int tid = threadIdx.x, wave = tid >> 5, lane = tid & 31;
  const int c0   = wave * 16;
  const int lrow = lane & 15;
  const int rsel = (lane >> 4) << 3;
  const int colL = c0 + lrow;

  for (int i = tid; i < Bn * HID; i += 512) (&h[0][0][0])[i] = (__bf16)0.0f;
  __syncthreads();

  for (int s = 0; s < Tn; ++s) {
    const int t   = dir ? (Tn - 1 - s) : s;
    const int cur = s & 1, nxt = cur ^ 1;
    const float* xgt = xg + (size_t)t * Bn * N3H;

    #pragma unroll 1
    for (int mp = 0; mp < 4; ++mp) {       // m-tile pairs: 2*mp, 2*mp+1
      v8f aR[2], aZ[2], aN[2];
      #pragma unroll
      for (int u = 0; u < 2; ++u) {
        const int mb = (mp * 2 + u) * 16;
        #pragma unroll
        for (int e = 0; e < 8; ++e) {
          int row = mb + e + rsel;
          aR[u][e] = xgt[(size_t)row * N3H + colL]           + bhh[colL];
          aZ[u][e] = xgt[(size_t)row * N3H + HID + colL]     + bhh[HID + colL];
          aN[u][e] = bhh[2 * HID + colL];
        }
      }
      #pragma unroll 1
      for (int kt = 0; kt < 8; ++kt) {     // K = 256 = 8 x 32
        const int kb = kt * 32 + rsel;
        v16bf bR = load_frag(Whh, HID, 0 * HID + c0 + lrow, kb);
        v16bf bZ = load_frag(Whh, HID, 1 * HID + c0 + lrow, kb);
        v16bf bN = load_frag(Whh, HID, 2 * HID + c0 + lrow, kb);
        v16bf a0 = load_frag(&h[cur][0][0], HID, (mp * 2 + 0) * 16 + lrow, kb);
        v16bf a1 = load_frag(&h[cur][0][0], HID, (mp * 2 + 1) * 16 + lrow, kb);
        aR[0] = wmma_bf16(a0, bR, aR[0]);
        aZ[0] = wmma_bf16(a0, bZ, aZ[0]);
        aN[0] = wmma_bf16(a0, bN, aN[0]);
        aR[1] = wmma_bf16(a1, bR, aR[1]);
        aZ[1] = wmma_bf16(a1, bZ, aZ[1]);
        aN[1] = wmma_bf16(a1, bN, aN[1]);
      }
      #pragma unroll
      for (int u = 0; u < 2; ++u) {
        const int mb = (mp * 2 + u) * 16;
        #pragma unroll
        for (int e = 0; e < 8; ++e) {
          int row = mb + e + rsel;
          float r  = 1.0f / (1.0f + __expf(-aR[u][e]));
          float z  = 1.0f / (1.0f + __expf(-aZ[u][e]));
          float xn = xgt[(size_t)row * N3H + 2 * HID + colL];
          float nv = tanhf(xn + r * aN[u][e]);
          float ho = (float)h[cur][row][colL];
          float hn = (1.0f - z) * nv + z * ho;
          h[nxt][row][colL] = (__bf16)hn;
          if (write_all)
            out[((size_t)t * Bn + row) * (2 * HID) + dir * HID + colL] = (__bf16)hn;
        }
      }
    }
    __syncthreads();
  }

  if (!write_all) {                        // final state sits in h[0]
    for (int i = tid; i < Bn * HID; i += 512) {
      int row = i >> 8, col = i & 255;
      out[(size_t)row * (2 * HID) + dir * HID + col] = h[0][row][col];
    }
  }
}

// ---------------------------------------------------------------------------
extern "C" void kernel_launch(void* const* d_in, const int* in_sizes, int n_in,
                              void* d_out, int out_size, void* d_ws, size_t ws_size,
                              hipStream_t stream) {
  (void)in_sizes; (void)n_in; (void)out_size; (void)ws_size;

  const int*   wi      = (const int*)  d_in[0];
  const float* embm    = (const float*)d_in[1];
  const float* oov     = (const float*)d_in[2];
  const float* Wih0f   = (const float*)d_in[3];
  const float* Whh0f   = (const float*)d_in[4];
  const float* bih0f   = (const float*)d_in[5];
  const float* bhh0f   = (const float*)d_in[6];
  const float* Wih0b   = (const float*)d_in[7];
  const float* Whh0b   = (const float*)d_in[8];
  const float* bih0b   = (const float*)d_in[9];
  const float* bhh0b   = (const float*)d_in[10];
  const float* Wih1f   = (const float*)d_in[11];
  const float* Whh1f   = (const float*)d_in[12];
  const float* bih1f   = (const float*)d_in[13];
  const float* bhh1f   = (const float*)d_in[14];
  const float* Wih1b   = (const float*)d_in[15];
  const float* Whh1b   = (const float*)d_in[16];
  const float* bih1b   = (const float*)d_in[17];
  const float* bhh1b   = (const float*)d_in[18];
  const float* projW   = (const float*)d_in[19];
  const float* projb   = (const float*)d_in[20];

  // ---- workspace carve-up (256B aligned) ----
  char* ws = (char*)d_ws;
  size_t off = 0;
  auto carve = [&](size_t bytes) {
    char* p = ws + off;
    off = (off + bytes + 255) & ~(size_t)255;
    return p;
  };
  const size_t MTOT = (size_t)Tn * Bn;                 // 8192 rows
  __bf16* x_bf    = (__bf16*)carve(MTOT * KP0 * 2);
  __bf16* w_ih0f  = (__bf16*)carve((size_t)N3H * KP0 * 2);
  __bf16* w_ih0b  = (__bf16*)carve((size_t)N3H * KP0 * 2);
  __bf16* w_hh0f  = (__bf16*)carve((size_t)N3H * HID * 2);
  __bf16* w_hh0b  = (__bf16*)carve((size_t)N3H * HID * 2);
  __bf16* w_hh1f  = (__bf16*)carve((size_t)N3H * HID * 2);
  __bf16* w_hh1b  = (__bf16*)carve((size_t)N3H * HID * 2);
  __bf16* w_ih1f  = (__bf16*)carve((size_t)N3H * K1  * 2);
  __bf16* w_ih1b  = (__bf16*)carve((size_t)N3H * K1  * 2);
  __bf16* w_proj  = (__bf16*)carve((size_t)HID * K1  * 2);
  float*  xga     = (float*) carve(MTOT * N3H * 4);    // xg0f, reused as xg1f
  float*  xgb     = (float*) carve(MTOT * N3H * 4);    // xg0b, reused as xg1b
  __bf16* x1      = (__bf16*)carve(MTOT * K1 * 2);     // concat(hf0, hb0)
  __bf16* fin     = (__bf16*)carve((size_t)Bn * K1 * 2);

  // 1) embedding gather + OOV + pad -> bf16
  embed_pad_kernel<<<(int)MTOT, KP0, 0, stream>>>(wi, embm, oov, x_bf);

  // 2) weight conversions
  auto cvt = [&](const float* s, __bf16* d, int N, int K, int KP) {
    int blocks = (int)(((size_t)N * KP + 255) / 256);
    cvt_pad_bf16_kernel<<<blocks, 256, 0, stream>>>(s, d, N, K, KP);
  };
  cvt(Wih0f, w_ih0f, N3H, EMB, KP0);
  cvt(Wih0b, w_ih0b, N3H, EMB, KP0);
  cvt(Whh0f, w_hh0f, N3H, HID, HID);
  cvt(Whh0b, w_hh0b, N3H, HID, HID);
  cvt(Whh1f, w_hh1f, N3H, HID, HID);
  cvt(Whh1b, w_hh1b, N3H, HID, HID);
  cvt(Wih1f, w_ih1f, N3H, K1, K1);
  cvt(Wih1b, w_ih1b, N3H, K1, K1);
  cvt(projW, w_proj, HID, K1, K1);

  // 3) layer-0 input projections: xg0 = x * Wih0^T + bih0   [8192 x 768]
  dim3 gXg(N3H / 128, (int)(MTOT / 128));
  gemm_bf16_nt_kernel<<<gXg, 256, 0, stream>>>(x_bf, w_ih0f, bih0f, xga, (int)MTOT, N3H, KP0);
  gemm_bf16_nt_kernel<<<gXg, 256, 0, stream>>>(x_bf, w_ih0b, bih0b, xgb, (int)MTOT, N3H, KP0);

  // 4) layer-0 recurrence (both directions concurrently), emits x1 bf16
  gru_scan_kernel<<<2, 512, 0, stream>>>(xga, xgb, w_hh0f, w_hh0b, bhh0f, bhh0b, x1, 1);

  // 5) layer-1 input projections (reuse xg buffers): xg1 = x1 * Wih1^T + bih1
  gemm_bf16_nt_kernel<<<gXg, 256, 0, stream>>>(x1, w_ih1f, bih1f, xga, (int)MTOT, N3H, K1);
  gemm_bf16_nt_kernel<<<gXg, 256, 0, stream>>>(x1, w_ih1b, bih1b, xgb, (int)MTOT, N3H, K1);

  // 6) layer-1 recurrence, emits boundary states into fin[128,512]
  gru_scan_kernel<<<2, 512, 0, stream>>>(xga, xgb, w_hh1f, w_hh1b, bhh1f, bhh1b, fin, 0);

  // 7) projection: out[128,256] = fin * projW^T + proj_b
  dim3 gP(HID / 128, Bn / 128);
  gemm_bf16_nt_kernel<<<gP, 256, 0, stream>>>(fin, w_proj, projb, (float*)d_out, Bn, HID, K1);
}
